// ScaleVQ_63866163692074
// MI455X (gfx1250) — matrix operand order repaired and checked by reading, not compiled, not run on hardware
//
#include <hip/hip_runtime.h>
#include <hip/hip_bf16.h>

typedef __attribute__((ext_vector_type(16))) __bf16 v16bf;
typedef __attribute__((ext_vector_type(8)))  float  v8f;
typedef __attribute__((ext_vector_type(4)))  unsigned int v4u;
typedef __attribute__((ext_vector_type(8)))  int    v8i;
typedef __attribute__((ext_vector_type(4)))  int    v4i;

typedef unsigned short ushort_t;
typedef unsigned int   uint_t;

union FragBF {
    uint4 u[2];
    v16bf v;
};

#define Bq   64
#define Cq   512
#define Tq   128
#define Qq   8
#define Vq   1024
#define Kq   3
#define NZ   (Bq * Cq * Tq)          // 4,194,304
#define KW   (Cq * Kq)               // 1536 (im2col K)
#define NIDX (Bq * 255)

#if defined(__has_builtin)
#if __has_builtin(__builtin_amdgcn_tensor_load_to_lds)
#define HAVE_TDM 1
#endif
#endif
#ifndef HAVE_TDM
#define HAVE_TDM 0
#endif

__device__ __forceinline__ ushort_t f2bf(float x) {
    uint_t u = __float_as_uint(x);
    uint_t r = u + 0x7FFFu + ((u >> 16) & 1u);   // round-to-nearest-even
    return (ushort_t)(r >> 16);
}

#if HAVE_TDM
// TDM: load a 128-row x 32-col bf16 tile from a (512 x 1536) row-major tensor
// into LDS, padding +4 DWORDs after every 16 DWORDs (row) -> 40-ushort row stride.
__device__ __forceinline__ void tdm_load_tile_a(const ushort_t* gsrc, ushort_t* lds_dst) {
    unsigned long long ga = (unsigned long long)(size_t)gsrc;
    unsigned lp = (unsigned)(size_t)lds_dst;     // LDS aperture: addr[31:0] is LDS offset
    v4u g0 = {
        1u,                                       // [1:0] count=1 (valid user descriptor)
        lp,                                       // [63:32] lds_addr
        (unsigned)ga,                             // global_addr[31:0]
        (unsigned)((ga >> 32) & 0x01FFFFFFull) | (2u << 30)  // global_addr[56:32] | type=2
    };
    v8i g1 = {
        (int)((1u << 16) | (1u << 20) | (3u << 22) | (3u << 25)),
        //    data_size=2B  pad_enable  pad_interval=16DW  pad_amount=4DW
        (int)(((unsigned)KW & 0xFFFFu) << 16),                        // tensor_dim0 lo16
        (int)((((unsigned)KW >> 16) & 0xFFFFu) | (((unsigned)Cq & 0xFFFFu) << 16)),
        (int)((((unsigned)Cq >> 16) & 0xFFFFu) | (32u << 16)),        // tile_dim0 = 32
        (int)(128u),                                                  // tile_dim1=128, tile_dim2=0
        (int)KW,                                                      // tensor_dim0_stride lo32
        0,                                                            // stride hi | dim1_stride lo
        0
    };
    v4i zz = {0, 0, 0, 0};
#if defined(__clang_major__) && (__clang_major__ >= 23)
    v8i z8 = {0, 0, 0, 0, 0, 0, 0, 0};
    __builtin_amdgcn_tensor_load_to_lds(g0, g1, zz, zz, z8, 0);
#else
    __builtin_amdgcn_tensor_load_to_lds(g0, g1, zz, zz, 0);
#endif
}
#endif

// ---------------- init: z_q = z, z_hat = 0, counts = 0, sse = 0 ----------------
__global__ void k_init(const float* __restrict__ z, float* __restrict__ z_q,
                       float* __restrict__ z_hat, float* __restrict__ counts,
                       float* __restrict__ sse) {
    long i = (long)blockIdx.x * blockDim.x + threadIdx.x;
    if (i < NZ) { z_q[i] = z[i]; z_hat[i] = 0.0f; }
    if (i < Qq * Vq) counts[i] = 0.0f;
    if (i < Qq) sse[i] = 0.0f;
}

// ---------------- f32 -> bf16 bulk convert ----------------
__global__ void k_tobf(const float* __restrict__ src, ushort_t* __restrict__ dst, long n) {
    long i = (long)blockIdx.x * blockDim.x + threadIdx.x;
    if (i < n) dst[i] = f2bf(src[i]);
}

// ---------------- codebook squared norms ----------------
__global__ void k_cbnorm(const float* __restrict__ cb, float* __restrict__ cbn) {
    int i = blockIdx.x * blockDim.x + threadIdx.x;      // (q,v) flat
    if (i >= Qq * Vq) return;
    const float* row = cb + (long)i * Cq;
    float s = 0.0f;
    for (int c = 0; c < Cq; ++c) s += row[c] * row[c];
    cbn[i] = s;
}

// ---------------- window-mean downsample, write bf16 rows (B*s, C) ----------------
__global__ void k_down(const float* __restrict__ z_q, ushort_t* __restrict__ xd, int s) {
    long i = (long)blockIdx.x * blockDim.x + threadIdx.x;
    long n = (long)Bq * s * Cq;
    if (i >= n) return;
    int c = (int)(i % Cq);
    long r = i / Cq;                 // (b, j) row
    int b = (int)(r / s), j = (int)(r % s);
    int w = Tq / s;
    const float* p = z_q + ((long)b * Cq + c) * Tq + (long)j * w;
    float sum = 0.0f;
    for (int e = 0; e < w; ++e) sum += p[e];
    xd[r * Cq + c] = f2bf(sum / (float)w);
}

// ---------------- VQ: bf16 WMMA GEMM (M=B*s, K=512, N=1024) fused with argmin ----------------
__global__ __launch_bounds__(256) void k_vq(
        const ushort_t* __restrict__ xd,     // (B*s, 512) bf16
        const ushort_t* __restrict__ cbq,    // (1024, 512) bf16 (this scale)
        const float*    __restrict__ cbn,    // (1024)
        int*            __restrict__ idx_out,
        float*          __restrict__ counts_q,
        float*          __restrict__ out_idx_f,   // d_out index region base
        int s, int qoff) {
    __shared__ float sVal[256][8];
    __shared__ int   sIdx[256][8];

    const int tid  = threadIdx.x;
    const int wave = tid >> 5, lane = tid & 31;
    const int lh = lane & 15, hi = lane >> 4;
    const int mblk = blockIdx.x;             // 16 rows per block

    float bestV[8];
    int   bestI[8];
#pragma unroll
    for (int r = 0; r < 8; ++r) { bestV[r] = 3.4e38f; bestI[r] = 0; }

    const ushort_t* arow = xd + ((long)mblk * 16 + lh) * Cq;

    for (int nt = 0; nt < 8; ++nt) {         // each wave covers 128 columns
        int v0 = wave * 128 + nt * 16;
        v8f acc = {};
        const ushort_t* brow = cbq + ((long)(v0 + lh)) * Cq + hi * 16;
#pragma unroll
        for (int kk = 0; kk < Cq; kk += 32) {
            FragBF a, bmat;
            a.u[0] = *(const uint4*)(arow + kk + hi * 16);
            a.u[1] = *(const uint4*)(arow + kk + hi * 16 + 8);
            bmat.u[0] = *(const uint4*)(brow + kk);
            bmat.u[1] = *(const uint4*)(brow + kk + 8);
            acc = __builtin_amdgcn_wmma_f32_16x16x32_bf16(
                false, a.v, false, bmat.v, (short)0, acc, false, false);
        }
        int v = v0 + lh;
        float cn = cbn[v];
#pragma unroll
        for (int r = 0; r < 8; ++r) {
            float sc = cn - 2.0f * acc[r];
            if (sc < bestV[r] || (sc == bestV[r] && v < bestI[r])) { bestV[r] = sc; bestI[r] = v; }
        }
    }

#pragma unroll
    for (int r = 0; r < 8; ++r) { sVal[tid][r] = bestV[r]; sIdx[tid][r] = bestI[r]; }
    __syncthreads();

    if (tid < 16) {
        int M = tid, r = M & 7, h = M >> 3;
        float bv = 3.4e38f; int bi = 0;
        for (int w2 = 0; w2 < 8; ++w2)
            for (int l2 = 0; l2 < 16; ++l2) {
                int t2 = w2 * 32 + h * 16 + l2;
                float vv = sVal[t2][r];
                int   ii = sIdx[t2][r];
                if (vv < bv || (vv == bv && ii < bi)) { bv = vv; bi = ii; }
            }
        int m = mblk * 16 + M;
        idx_out[m] = bi;
        atomicAdd(&counts_q[bi], 1.0f);
        int b = m / s, j = m % s;
        out_idx_f[(long)b * 255 + qoff + j] = (float)bi;
    }
}

// ---------------- gather + linear upsample -> bf16 (B,C,T) ----------------
__global__ void k_up(const float* __restrict__ cbq, const int* __restrict__ idx,
                     ushort_t* __restrict__ up, int s) {
    long i = (long)blockIdx.x * blockDim.x + threadIdx.x;
    if (i >= NZ) return;
    int t = (int)(i & (Tq - 1));
    int c = (int)((i >> 7) & (Cq - 1));
    int b = (int)(i >> 16);
    float pos = (t + 0.5f) * (float)s * (1.0f / (float)Tq) - 0.5f;
    float fl = floorf(pos);
    int i0 = (int)fl;
    float f = pos - fl;
    int i0c = i0 < 0 ? 0 : (i0 > s - 1 ? s - 1 : i0);
    int i1c = i0 + 1 < 0 ? 0 : (i0 + 1 > s - 1 ? s - 1 : i0 + 1);
    float v0 = cbq[(long)idx[b * s + i0c] * Cq + c];
    float v1 = cbq[(long)idx[b * s + i1c] * Cq + c];
    up[i] = f2bf(v0 + f * (v1 - v0));
}

// ---------------- Conv1d(K=3) + bias + ReLU as bf16 WMMA GEMM ----------------
// MODE 0: out -> bf16 buffer. MODE 1: z_hat += y; z_q -= y; sse += (z_hat - z)^2
template <int MODE>
__global__ __launch_bounds__(256) void k_conv(
        const ushort_t* __restrict__ xin,    // (B, C, T) bf16
        const ushort_t* __restrict__ wbf,    // (512, 1536) bf16 row-major
        const float*    __restrict__ bias,   // (512)
        ushort_t*       __restrict__ obf,    // MODE 0
        float* __restrict__ z_hat, float* __restrict__ z_q,
        const float* __restrict__ z, float* __restrict__ sse) {
    __shared__ ushort_t lds_a[128 * 40];     // weights tile (m-local, kk-local), stride 40
    __shared__ ushort_t lds_b[128 * 40];     // im2col, t-major (t, kk-local)
    __shared__ float    red[256];

    const int b  = blockIdx.y;
    const int m0 = blockIdx.x * 128;         // 128 output channels per block
    const int tid  = threadIdx.x;
    const int wave = tid >> 5, lane = tid & 31;
    const int lh = lane & 15, hi = lane >> 4;

    const ushort_t* xb = xin + (long)b * Cq * Tq;

    v8f acc[8] = {};

    for (int kk0 = 0; kk0 < KW; kk0 += 32) {
        // ---- stage A (weights 128x32 tile) via Tensor Data Mover from wave 0 ----
#if HAVE_TDM
        if (tid < 32) {
            tdm_load_tile_a(wbf + (long)m0 * KW + kk0, lds_a);
        }
#else
        {
            int m = tid >> 1, half = tid & 1;
            const ushort_t* src = wbf + ((long)(m0 + m)) * KW + kk0 + half * 16;
            uint4 p0 = *(const uint4*)(src);
            uint4 p1 = *(const uint4*)(src + 8);
            ushort_t* dst = &lds_a[m * 40 + half * 16];
            *(uint4*)(dst)     = p0;
            *(uint4*)(dst + 8) = p1;
        }
#endif
        // ---- stage B: im2col rows kk0..kk0+31 over all t, transposed ----
        {
            int kkl = tid >> 3, seg = tid & 7;
            int kk  = kk0 + kkl;
            int cin = kk / 3, k = kk - cin * 3;
            int sh  = k - 1;
            const ushort_t* srow = xb + (long)cin * Tq;
            if (kk0 + 32 < KW)
                __builtin_prefetch(srow + ((kk + 32) / 3 - cin) * Tq, 0, 1);
#pragma unroll
            for (int e = 0; e < 16; ++e) {
                int t  = seg * 16 + e;
                int ts = t + sh;
                ushort_t val = (ts >= 0 && ts < Tq) ? srow[ts] : (ushort_t)0;
                lds_b[t * 40 + kkl] = val;
            }
        }
#if HAVE_TDM
        if (tid < 32) __builtin_amdgcn_s_wait_tensorcnt(0);
#endif
        __syncthreads();

        // ---- compute: each wave owns a 16-row M strip x all 128 t ----
        FragBF a;
        const ushort_t* ap = &lds_a[(wave * 16 + lh) * 40];
        a.u[0] = *(const uint4*)(ap + hi * 16);
        a.u[1] = *(const uint4*)(ap + hi * 16 + 8);
#pragma unroll
        for (int nt = 0; nt < 8; ++nt) {
            FragBF bb;
            const ushort_t* bp = &lds_b[(nt * 16 + lh) * 40 + hi * 16];
            bb.u[0] = *(const uint4*)(bp);
            bb.u[1] = *(const uint4*)(bp + 8);
            acc[nt] = __builtin_amdgcn_wmma_f32_16x16x32_bf16(
                false, a.v, false, bb.v, (short)0, acc[nt], false, false);
        }
        __syncthreads();
    }

    // ---- epilogue ----
    float bs[8];
#pragma unroll
    for (int r = 0; r < 8; ++r) bs[r] = bias[m0 + wave * 16 + r + 8 * hi];

    float local_sse = 0.0f;
#pragma unroll
    for (int nt = 0; nt < 8; ++nt) {
        int t = nt * 16 + lh;
#pragma unroll
        for (int r = 0; r < 8; ++r) {
            int cout = m0 + wave * 16 + r + 8 * hi;
            float y = acc[nt][r] + bs[r];
            y = y > 0.0f ? y : 0.0f;
            long off = ((long)b * Cq + cout) * Tq + t;
            if (MODE == 0) {
                obf[off] = f2bf(y);
            } else {
                float zh = z_hat[off] + y;
                z_hat[off] = zh;
                z_q[off]  -= y;
                float d = zh - z[off];
                local_sse += d * d;
            }
        }
    }
    if (MODE == 1) {
        red[tid] = local_sse;
        __syncthreads();
        for (int st = 128; st > 0; st >>= 1) {
            if (tid < st) red[tid] += red[tid + st];
            __syncthreads();
        }
        if (tid == 0) atomicAdd(sse, red[0]);
    }
}

// ---------------- finalize scalars: usage, loss, perplexity ----------------
__global__ __launch_bounds__(256) void k_final(const float* __restrict__ counts,
                                               const float* __restrict__ sse,
                                               float* __restrict__ out3) {
    __shared__ float red[256];
    const int tid = threadIdx.x;
    float usage_acc = 0.0f, perp_acc = 0.0f;
    for (int q = 0; q < Qq; ++q) {
        int s = 1 << q;
        float denom = (float)(Bq * s);
        float lu = 0.0f, lp = 0.0f;
        for (int v = tid; v < Vq; v += 256) {
            float cnt = counts[q * Vq + v];
            lu += (cnt > 0.0f) ? 1.0f : 0.0f;
            float p = cnt / denom;
            lp += p * logf(p + 1e-10f);
        }
        red[tid] = lu; __syncthreads();
        for (int st = 128; st > 0; st >>= 1) { if (tid < st) red[tid] += red[tid + st]; __syncthreads(); }
        if (tid == 0) usage_acc += red[0] * (100.0f / (float)Vq);
        __syncthreads();
        red[tid] = lp; __syncthreads();
        for (int st = 128; st > 0; st >>= 1) { if (tid < st) red[tid] += red[tid + st]; __syncthreads(); }
        if (tid == 0) perp_acc += expf(-red[0]);
        __syncthreads();
    }
    if (tid == 0) {
        float loss_acc = 0.0f;
        for (int q = 0; q < Qq; ++q)
            loss_acc += 1.25f * sse[q] / (float)NZ;
        out3[0] = usage_acc / (float)Qq;   // usage
        out3[1] = loss_acc  / (float)Qq;   // loss
        out3[2] = perp_acc  / (float)Qq;   // perplexity
    }
}

// ---------------- z_hat -> d_out (straight-through output == z_hat) ----------------
__global__ void k_copy(const float* __restrict__ z_hat, float* __restrict__ out) {
    long i = (long)blockIdx.x * blockDim.x + threadIdx.x;
    if (i < NZ) out[i] = z_hat[i];
}

extern "C" void kernel_launch(void* const* d_in, const int* in_sizes, int n_in,
                              void* d_out, int out_size, void* d_ws, size_t ws_size,
                              hipStream_t stream) {
    (void)in_sizes; (void)n_in; (void)out_size; (void)ws_size;
    const float* z  = (const float*)d_in[0];   // (B,C,T)
    const float* cb = (const float*)d_in[1];   // (Q,V,C)
    const float* pw = (const float*)d_in[2];   // (Q,DEPTH,C,C,K)
    const float* pb = (const float*)d_in[3];   // (Q,DEPTH,C)
    float* out = (float*)d_out;

    char* ws = (char*)d_ws;
    float*    z_q    = (float*)ws;     ws += (size_t)NZ * 4;              // 16 MB
    float*    z_hat  = (float*)ws;     ws += (size_t)NZ * 4;              // 16 MB
    ushort_t* up_bf  = (ushort_t*)ws;  ws += (size_t)NZ * 2;              // 8 MB
    ushort_t* x1_bf  = (ushort_t*)ws;  ws += (size_t)NZ * 2;              // 8 MB
    ushort_t* xd_bf  = (ushort_t*)ws;  ws += (size_t)Bq * Tq * Cq * 2;    // 8 MB
    ushort_t* w_bf   = (ushort_t*)ws;  ws += (size_t)Qq * 2 * Cq * KW * 2;// 24 MB
    ushort_t* cb_bf  = (ushort_t*)ws;  ws += (size_t)Qq * Vq * Cq * 2;    // 8 MB
    float*    cbnorm = (float*)ws;     ws += (size_t)Qq * Vq * 4;
    float*    counts = (float*)ws;     ws += (size_t)Qq * Vq * 4;
    float*    sse    = (float*)ws;     ws += 256;
    int*      idx    = (int*)ws;       ws += (size_t)Bq * Tq * 4;

    const long NW = (long)Qq * 2 * Cq * KW;     // weight elements
    const long NC = (long)Qq * Vq * Cq;         // codebook elements

    k_init<<<(NZ + 255) / 256, 256, 0, stream>>>(z, z_q, z_hat, counts, sse);
    k_tobf<<<(int)((NW + 255) / 256), 256, 0, stream>>>(pw, w_bf, NW);
    k_tobf<<<(int)((NC + 255) / 256), 256, 0, stream>>>(cb, cb_bf, NC);
    k_cbnorm<<<(Qq * Vq + 255) / 256, 256, 0, stream>>>(cb, cbnorm);

    int qoff = 0;
    for (int q = 0; q < Qq; ++q) {
        int s = 1 << q;
        long ndown = (long)Bq * s * Cq;
        k_down<<<(int)((ndown + 255) / 256), 256, 0, stream>>>(z_q, xd_bf, s);
        k_vq<<<dim3(Bq * s / 16), 256, 0, stream>>>(
            xd_bf, cb_bf + (size_t)q * Vq * Cq, cbnorm + q * Vq,
            idx, counts + q * Vq, out + NZ + 3, s, qoff);
        k_up<<<(NZ + 255) / 256, 256, 0, stream>>>(cb + (size_t)q * Vq * Cq, idx, up_bf, s);
        k_conv<0><<<dim3(Cq / 128, Bq), 256, 0, stream>>>(
            up_bf, w_bf + (size_t)(q * 2 + 0) * Cq * KW, pb + (q * 2 + 0) * Cq,
            x1_bf, nullptr, nullptr, nullptr, nullptr);
        k_conv<1><<<dim3(Cq / 128, Bq), 256, 0, stream>>>(
            x1_bf, w_bf + (size_t)(q * 2 + 1) * Cq * KW, pb + (q * 2 + 1) * Cq,
            nullptr, z_hat, z_q, z, sse + q);
        qoff += s;
    }

    k_final<<<1, 256, 0, stream>>>(counts, sse, out + NZ);
    k_copy<<<(NZ + 255) / 256, 256, 0, stream>>>(z_hat, out);
}
